// Individual_23648089932011
// MI455X (gfx1250) — compile-verified, matrix operand order checked by reading, not run on hardware
//
#include <hip/hip_runtime.h>
#include <hip/hip_bf16.h>

// ---------------------------------------------------------------------------
// Problem constants (from the reference)
// ---------------------------------------------------------------------------
#define BATCH_   65536
#define IN_      256      // input columns of buf
#define MID_     272      // middle nodes (17 blocks of 16)
#define NODES_   528
#define OUT_     16
#define NBLK     17       // 272 / 16

// Layout constants
#define KPW      544      // row stride (halfs) of transposed masked weights (528 -> pad 544)
#define KPS      536      // LDS activation row stride in halfs (4*67 dwords -> conflict-free)
#define TILE_M   32       // batch rows per workgroup (2 waves x 16 rows)
#define BLOCK_T  64       // threads per workgroup

typedef __attribute__((ext_vector_type(16))) _Float16 v16h;
typedef __attribute__((ext_vector_type(8)))  float    v8f;

// 16-bit A/B fragment (16x32 tile) loader. CDNA5 layout (ISA 7.12.2):
//   lanes 0-15 : row = lane,   elements 0..7 = K 0..7,  elements 8..15 = K 16..23
//   lanes16-31 : row = lane-16,elements 0..7 = K 8..15, elements 8..15 = K 24..31
// Caller passes p = &row[k0 + (lane>=16 ? 8 : 0)]; we grab two 16B chunks.
__device__ __forceinline__ v16h load_frag(const _Float16* p) {
    union { v16h h; struct { uint4 lo; uint4 hi; } q; } u;
    u.q.lo = *(const uint4*)(p);
    u.q.hi = *(const uint4*)(p + 16);
    return u.h;
}

__device__ __forceinline__ float fsig(float x) {
    return __builtin_amdgcn_rcpf(1.0f + __expf(-x));
}

// wave-uniform lane broadcast without LDS traffic: v_readlane_b32
__device__ __forceinline__ float bcast_lane(float v, int srcLane) {
    return __uint_as_float(__builtin_amdgcn_readlane(__float_as_uint(v), srcLane));
}

// ---------------------------------------------------------------------------
// Prep: build (a) WTh[j][k] = f16( weight[k][j] * conn[k][j] )   j<272, k<528
//             row-major, row stride KPW, zero-padded  (WMMA B operand source)
//       (b) Wmid[j][n] = f32( weight[256+j][n] * conn[256+j][n] ) (rank-1 rows)
// ---------------------------------------------------------------------------
__global__ void ng_prep(const float* __restrict__ w, const int* __restrict__ conn,
                        _Float16* __restrict__ WTh, float* __restrict__ Wmid) {
    int idx = blockIdx.x * 256 + threadIdx.x;
    if (idx < MID_ * KPW) {
        int j = idx / KPW, k = idx % KPW;
        _Float16 v = (_Float16)0.0f;
        if (k < NODES_) v = (_Float16)(w[(size_t)k * MID_ + j] * (float)conn[(size_t)k * MID_ + j]);
        WTh[idx] = v;
    }
    if (idx < MID_ * MID_) {
        int j = idx / MID_, n = idx % MID_;
        size_t o = (size_t)(IN_ + j) * MID_ + n;
        Wmid[idx] = w[o] * (float)conn[o];
    }
}

// ---------------------------------------------------------------------------
// Main: blocked triangular recurrence. Each wave owns 16 batch rows.
// ---------------------------------------------------------------------------
__global__ __launch_bounds__(BLOCK_T)
void ng_forward(const float*    __restrict__ x,
                const _Float16* __restrict__ WTh,
                const float*    __restrict__ Wmid,
                const float*    __restrict__ bias,
                const int*      __restrict__ exist,
                float*          __restrict__ out) {
    __shared__ __align__(16) _Float16 S[TILE_M * KPS];   // activations (f16)
    __shared__ float sBias[MID_];
    __shared__ float sExist[MID_];

    const int tid     = threadIdx.x;
    const int rowBase = blockIdx.x * TILE_M;

    // x tile -> f16 into S cols [0,256)
    for (int i = tid; i < TILE_M * (IN_ / 8); i += BLOCK_T) {
        int r = i / (IN_ / 8), c = (i % (IN_ / 8)) * 8;
        const float4* src = (const float4*)(x + (size_t)(rowBase + r) * IN_ + c);
        float4 v0 = src[0], v1 = src[1];
        union { _Float16 h[8]; uint4 u; } pk;
        pk.h[0] = (_Float16)v0.x; pk.h[1] = (_Float16)v0.y;
        pk.h[2] = (_Float16)v0.z; pk.h[3] = (_Float16)v0.w;
        pk.h[4] = (_Float16)v1.x; pk.h[5] = (_Float16)v1.y;
        pk.h[6] = (_Float16)v1.z; pk.h[7] = (_Float16)v1.w;
        *(uint4*)&S[r * KPS + c] = pk.u;
    }
    // zero middle-node columns [256, KPS): un-computed nodes must read as 0
    for (int i = tid; i < TILE_M * ((KPS - IN_) / 4); i += BLOCK_T) {
        int r = i / ((KPS - IN_) / 4), c = (i % ((KPS - IN_) / 4)) * 4;
        *(uint2*)&S[r * KPS + IN_ + c] = make_uint2(0u, 0u);
    }
    for (int i = tid; i < MID_; i += BLOCK_T) {
        sBias[i]  = bias[i];
        sExist[i] = (float)exist[i];
    }
    __syncthreads();   // after this, each wave is fully independent

    const int wave = tid >> 5;
    const int lane = tid & 31;
    const int m    = lane & 15;             // row-within-tile (A) / node-within-block (B)
    const int kh   = (lane >> 4) << 3;      // K phase: 0 or 8

    const _Float16* aRow = &S[(wave * 16 + m) * KPS];

    for (int b = 0; b < NBLK; ++b) {
        const int nodeBase = 16 * b;
        const int Kend  = IN_ + nodeBase;          // valid sources for this block
        const int Kceil = (Kend + 31) & ~31;       // round up; extra cols are zero
        const _Float16* bRow = WTh + (size_t)(nodeBase + m) * KPW;

        // acc[M=16, N=16] += S[M, 0:Kceil] @ W[0:Kceil, block]
        v8f acc = {};
        for (int k0 = 0; k0 < Kceil; k0 += 32) {
            v16h af = load_frag(aRow + k0 + kh);
            v16h bf = load_frag(bRow + k0 + kh);
            if (k0 + 32 < Kceil)
                __builtin_prefetch(bRow + k0 + 32 + kh, 0, 1);  // global_prefetch_b8
            acc = __builtin_amdgcn_wmma_f32_16x16x32_f16(
                      false, af, false, bf, (short)0, acc, false, false);
        }

        // Hoist all memory off the serial chain: rank-1 weights, bias, exist.
        float wcol[16], bcol[16], ecol[16];
        #pragma unroll
        for (int jj = 0; jj < 16; ++jj) {
            wcol[jj] = Wmid[(size_t)(nodeBase + jj) * MID_ + nodeBase + m];
            bcol[jj] = sBias[nodeBase + jj];
            ecol[jj] = sExist[nodeBase + jj];
        }

        // 16-step sequential recurrence inside the block.
        // C-tile layout: VGPR v holds rows {v, v+8}; column n lives in lanes {n, n+16}.
        #pragma unroll 1
        for (int jj = 0; jj < 16; ++jj) {
            const int   j  = nodeBase + jj;
            const float bi = bcol[jj];
            const float ex = ecol[jj];
            const float wv = wcol[jj];
            #pragma unroll
            for (int v = 0; v < 8; ++v) {
                float tv  = (fsig(acc[v]) + bi) * ex;     // vector transform
                float hlo = bcast_lane(tv, jj);           // row v,   col jj (v_readlane)
                float hhi = bcast_lane(tv, jj + 16);      // row v+8, col jj
                float h   = (lane < 16) ? hlo : hhi;
                // update every column: cols < jj already consumed (harmless),
                // diagonal term added after consumption == reference's zero.
                acc[v] += h * wv;
                if (b < NBLK - 1) {
                    if (lane == 0) {                      // publish h into S (f16)
                        S[(wave * 16 + v)     * KPS + IN_ + j] = (_Float16)hlo;
                        S[(wave * 16 + v + 8) * KPS + IN_ + j] = (_Float16)hhi;
                    }
                } else {                                  // last block == the 16 outputs
                    if (lane == 0) {
                        out[(size_t)(rowBase + wave * 16 + v)     * OUT_ + jj] = hlo;
                        out[(size_t)(rowBase + wave * 16 + v + 8) * OUT_ + jj] = hhi;
                    }
                }
            }
        }
    }
}

// ---------------------------------------------------------------------------
extern "C" void kernel_launch(void* const* d_in, const int* in_sizes, int n_in,
                              void* d_out, int out_size, void* d_ws, size_t ws_size,
                              hipStream_t stream) {
    const float* x      = (const float*)d_in[0];
    const float* weight = (const float*)d_in[1];
    const float* bias   = (const float*)d_in[2];
    const int*   conn   = (const int*)d_in[3];
    const int*   exist  = (const int*)d_in[4];
    float*       out    = (float*)d_out;

    // workspace: WTh (272*544 f16 = 295936 B) then Wmid (272*272 f32 = 295936 B)
    _Float16* WTh  = (_Float16*)d_ws;
    float*    Wmid = (float*)((char*)d_ws + (size_t)MID_ * KPW * sizeof(_Float16));

    const int prepElems = MID_ * KPW;   // covers both ranges (>= MID_*MID_)
    ng_prep<<<(prepElems + 255) / 256, 256, 0, stream>>>(weight, conn, WTh, Wmid);

    ng_forward<<<BATCH_ / TILE_M, BLOCK_T, 0, stream>>>(x, WTh, Wmid, bias, exist, out);
}
